// TransSSM_63239098466851
// MI455X (gfx1250) — compile-verified, matrix operand order checked by reading, not run on hardware
//
#include <hip/hip_runtime.h>
#include <cstdint>

// ---------------------------------------------------------------------------
// TransSSM on MI455X (gfx1250): bf16 WMMA GEMMs + wave32 selective scan.
// B=4, C=192, H=W=64, L=4096, N_STATE=16, DT_RANK=12, D_CONV=3, GROUPS=4.
//
// Direction handling (4 mamba passes):
//   dir bit0 (v)  = channel flip  -> folded into in_w column-reversal and
//                                    out_w row-reversal (weights only).
//   dir bit1 (sf) = sequence flip -> anti-causal depthwise conv + reversed
//                                    scan order; activations stay at original
//                                    positions so outputs need no re-flip.
//
// Round-2 changes:
//   * All GEMM operand staging now uses GLOBAL_LOAD_ASYNC_TO_LDS_B128
//     (ASYNCcnt path) instead of load->wait->ds_store round trips.
//   * Weight matrices are prepped in [N][K] (n-major, k-contiguous) layout so
//     B tiles load with the same 16B async chunks as A (no scalar transposes).
// ---------------------------------------------------------------------------

#define Bsz   4
#define Cch   192
#define Lseq  4096
#define Mrow  (Bsz * Lseq)        // 16384
#define NST   16
#define DTR   12
#define XDBL  48                  // 44 padded to 48 (3 x 16 WMMA tiles)

typedef __attribute__((ext_vector_type(16))) __bf16 v16bf;
typedef __attribute__((ext_vector_type(8)))  float  v8f;

__device__ __forceinline__ float bf2f(uint16_t u) {
  return __uint_as_float(((uint32_t)u) << 16);
}
__device__ __forceinline__ uint16_t f2bf(float f) {
  uint32_t u = __float_as_uint(f);
  uint32_t r = u + 0x7FFFu + ((u >> 16) & 1u);   // round-to-nearest-even
  return (uint16_t)(r >> 16);
}
__device__ __forceinline__ float sigmoidf_(float x) { return 1.f / (1.f + __expf(-x)); }
__device__ __forceinline__ float siluf_(float x)    { return x * sigmoidf_(x); }
__device__ __forceinline__ float softplusf_(float x){ return (x > 20.f) ? x : log1pf(__expf(x)); }

// CDNA5 async global->LDS 16-byte copy (per active lane), ASYNCcnt-tracked.
// LDS byte address = low 32 bits of the generic pointer (aperture rule).
__device__ __forceinline__ void async_lds_b128(void* lds, const void* g) {
  uint32_t l = (uint32_t)(uintptr_t)lds;
  uint64_t ga = (uint64_t)(uintptr_t)g;
  asm volatile("global_load_async_to_lds_b128 %0, %1, off" :: "v"(l), "v"(ga) : "memory");
}
__device__ __forceinline__ void wait_async0() {
  asm volatile("s_wait_asynccnt 0x0" ::: "memory");
}

// ---------------------------------------------------------------------------
// Weight prep: bf16, [N][K] layout (rows = output col, k contiguous), plus
// direction-flip variants folded in.
// ---------------------------------------------------------------------------
__global__ void prep_weights(const float* __restrict__ conv_w,  // [192][192] (o,c)
                             const float* __restrict__ in_w,    // [384][192] (e,c)
                             const float* __restrict__ xproj_w, // [44][192]
                             const float* __restrict__ out_w,   // [192][192] (o,c)
                             uint16_t* __restrict__ convW,      // [o=192][k=192]
                             uint16_t* __restrict__ inW,        // [n=768][k=192], n=v*384+e
                             uint16_t* __restrict__ xprojW,     // [n=48][k=192]
                             uint16_t* __restrict__ outW)       // [2][o=192][k=192]
{
  int tid = threadIdx.x + blockIdx.x * blockDim.x;
  int stride = blockDim.x * gridDim.x;
  for (int i = tid; i < Cch * Cch; i += stride)
    convW[i] = f2bf(conv_w[i]);                       // already (o,c) k-contiguous
  for (int i = tid; i < 768 * Cch; i += stride) {
    int n = i / Cch, c = i % Cch;
    int v = n / 384, e = n % 384;
    int cs = v ? (Cch - 1 - c) : c;                   // channel-flip variant
    inW[i] = f2bf(in_w[e * Cch + cs]);
  }
  for (int i = tid; i < XDBL * Cch; i += stride) {
    int n = i / Cch;
    xprojW[i] = f2bf(n < (DTR + 2 * NST) ? xproj_w[i] : 0.f);
  }
  for (int i = tid; i < 2 * Cch * Cch; i += stride) {
    int v = i / (Cch * Cch), r = i % (Cch * Cch);
    int o = r / Cch, c = r % Cch;
    int os = v ? (Cch - 1 - o) : o;                   // output-channel flip
    outW[i] = f2bf(out_w[os * Cch + c]);
  }
}

// ---------------------------------------------------------------------------
// ECA: mean pool over HW, conv7 along C, sigmoid -> per (b,c) scale.
// ---------------------------------------------------------------------------
__global__ void eca_pool(const float* __restrict__ x, float* __restrict__ pooled) {
  int bc = blockIdx.x;                     // 0..767
  const float* p = x + (size_t)bc * Lseq;
  float s = 0.f;
  for (int i = threadIdx.x; i < Lseq; i += blockDim.x) s += p[i];
  for (int m = 16; m >= 1; m >>= 1) s += __shfl_xor(s, m, 32);
  __shared__ float red[8];
  int wv = threadIdx.x >> 5, ln = threadIdx.x & 31;
  if (ln == 0) red[wv] = s;
  __syncthreads();
  if (threadIdx.x == 0) {
    float t = 0.f;
    for (int i = 0; i < 8; i++) t += red[i];
    pooled[bc] = t / (float)Lseq;
  }
}

__global__ void eca_sig(const float* __restrict__ pooled, const float* __restrict__ eca_w,
                        float* __restrict__ s_eca) {
  int i = threadIdx.x + blockIdx.x * blockDim.x;
  if (i >= Bsz * Cch) return;
  int b = i / Cch, c = i % Cch;
  float acc = 0.f;
  for (int t = 0; t < 7; t++) {
    int cc = c - 3 + t;
    if (cc >= 0 && cc < Cch) acc += pooled[b * Cch + cc] * eca_w[t];
  }
  s_eca[i] = sigmoidf_(acc);
}

// ---------------------------------------------------------------------------
// (B,C,L) f32 -> (B,L,C) bf16 via LDS tile transpose (GEMM A operand).
// ---------------------------------------------------------------------------
__global__ void transpose_bf16(const float* __restrict__ x, uint16_t* __restrict__ x_t) {
  __shared__ float tile[32][33];
  int l0 = blockIdx.x * 32, c0 = blockIdx.y * 32, b = blockIdx.z;
  const float* xb = x + (size_t)b * Cch * Lseq;
  for (int i = 0; i < 4; i++) {
    int c = c0 + threadIdx.y + i * 8;
    tile[threadIdx.y + i * 8][threadIdx.x] = xb[(size_t)c * Lseq + l0 + threadIdx.x];
  }
  __syncthreads();
  uint16_t* xt = x_t + (size_t)b * Lseq * Cch;
  for (int i = 0; i < 4; i++) {
    int l = l0 + threadIdx.y + i * 8;
    xt[(size_t)l * Cch + c0 + threadIdx.x] = f2bf(tile[threadIdx.x][threadIdx.y + i * 8]);
  }
}

// ---------------------------------------------------------------------------
// Generic bf16 WMMA GEMM: block = 128 thr (4 waves), tile 64 x (NT*16).
// A: row-major [M][lda] bf16.  B: row-major [N][ldb] bf16 (k contiguous).
// NSEG independent (A,B) pairs accumulated into one output tile.
// All staging via GLOBAL_LOAD_ASYNC_TO_LDS_B128.
// ---------------------------------------------------------------------------
struct Segs4 { const uint16_t* A[4]; const uint16_t* B[4]; };

union FragBf { v16bf v; uint32_t u[8]; };

template <int NT, int NSEG, class Epi>
__global__ __launch_bounds__(128) void gemm_bf16_k(Segs4 segs, int lda, int ldb, int K, Epi epi) {
  __shared__ alignas(16) uint16_t As[64][32];
  __shared__ alignas(16) uint16_t Bs[NT * 16][32];
  const int tid = threadIdx.x;
  const int wave = tid >> 5, lane = tid & 31;
  const int m0 = blockIdx.x * 64;
  const int n0 = blockIdx.y * (NT * 16);

  v8f acc[NT];
  for (int t = 0; t < NT; t++)
    for (int r = 0; r < 8; r++) acc[t][r] = 0.f;

  for (int s = 0; s < NSEG; s++) {
    const uint16_t* A = segs.A[s];
    const uint16_t* B = segs.B[s];
    for (int k0 = 0; k0 < K; k0 += 32) {
      // A tile 64x32: 256 async 16B chunks (8 bf16 each)
      for (int idx = tid; idx < 256; idx += 128) {
        int row = idx >> 2;
        int kc = (idx & 3) * 8;
        async_lds_b128(&As[row][kc], &A[(size_t)(m0 + row) * lda + k0 + kc]);
      }
      // B tile (NT*16)x32: NT*64 async chunks, same pattern (B is [N][K])
      for (int idx = tid; idx < NT * 64; idx += 128) {
        int row = idx >> 2;
        int kc = (idx & 3) * 8;
        async_lds_b128(&Bs[row][kc], &B[(size_t)(n0 + row) * ldb + k0 + kc]);
      }
      wait_async0();
      __syncthreads();

      // ISA 16-bit A 16x32 layout: lane%16 = row; lane/16 selects K-halves.
      int mr = wave * 16 + (lane & 15);
      int kh = (lane >> 4) * 8;
      FragBf fa;
      for (int i = 0; i < 4; i++) fa.u[i]     = *(const uint32_t*)&As[mr][kh + 2 * i];
      for (int i = 0; i < 4; i++) fa.u[4 + i] = *(const uint32_t*)&As[mr][16 + kh + 2 * i];

      for (int t = 0; t < NT; t++) {
        int nc = t * 16 + (lane & 15);
        FragBf fb;
        for (int i = 0; i < 4; i++) fb.u[i]     = *(const uint32_t*)&Bs[nc][kh + 2 * i];
        for (int i = 0; i < 4; i++) fb.u[4 + i] = *(const uint32_t*)&Bs[nc][16 + kh + 2 * i];
        acc[t] = __builtin_amdgcn_wmma_f32_16x16x32_bf16(
            false, fa.v, false, fb.v, (short)0, acc[t], false, false);
      }
      __syncthreads();
    }
  }

  // C/D layout: lane%16 = n, (lane/16)*8 + vgpr = m
  int mbase = m0 + wave * 16 + (lane >> 4) * 8;
  for (int t = 0; t < NT; t++) {
    int n = n0 + t * 16 + (lane & 15);
    for (int r = 0; r < 8; r++) epi.store(mbase + r, n, acc[t][r]);
  }
}

// Epilogues -----------------------------------------------------------------
struct EpiBiasBf {     // conv1x1: + bias, store bf16
  const float* bias; uint16_t* out; int ldo;
  __device__ void store(int m, int n, float v) const {
    out[(size_t)m * ldo + n] = f2bf(v + bias[n]);
  }
};
struct EpiBf {         // in_proj: store bf16
  uint16_t* out; int ldo;
  __device__ void store(int m, int n, float v) const {
    out[(size_t)m * ldo + n] = f2bf(v);
  }
};
struct EpiF32 {        // x_proj: store f32
  float* out; int ldo;
  __device__ void store(int m, int n, float v) const {
    out[(size_t)m * ldo + n] = v;
  }
};
struct EpiFinal {      // out_proj: /4 + ECA residual + channel shuffle
  const float* x; const float* s_eca; float* out;
  __device__ void store(int m, int n, float v) const {
    int b = m >> 12, l = m & (Lseq - 1);
    float x0 = x[((size_t)b * Cch + n) * Lseq + l] * s_eca[b * Cch + n];
    int j = (n % 48) * 4 + (n / 48);            // shuffle: out[b,j] = pre[b,(j%4)*48+j/4]
    out[((size_t)b * Cch + j) * Lseq + l] = v * 0.25f + x0;
  }
};

// ---------------------------------------------------------------------------
// Depthwise conv (causal / anti-causal) + SiLU.
// xz: [m][v*384 + e] bf16 (e<192 = x-part, e>=192 = z-part).
// ---------------------------------------------------------------------------
__global__ void dwconv_silu(const uint16_t* __restrict__ xz,
                            const float* __restrict__ mconv_w,  // [192][1][3]
                            const float* __restrict__ mconv_b,
                            uint16_t* __restrict__ xi_bf) {     // [4][M][192]
  size_t i = (size_t)blockIdx.x * blockDim.x + threadIdx.x;
  const size_t TOT = (size_t)4 * Mrow * Cch;
  if (i >= TOT) return;
  int d = (int)(i % Cch);
  size_t t = i / Cch;
  int m = (int)(t % Mrow);
  int dir = (int)(t / Mrow);
  int v = dir & 1, sf = dir >> 1;
  int b = m >> 12, l = m & (Lseq - 1);
  float acc = 0.f;
  if (!sf) {  // causal: x[l-2]w0 + x[l-1]w1 + x[l]w2
    for (int tt = 0; tt < 3; tt++) {
      int ll = l - 2 + tt;
      if (ll >= 0)
        acc += bf2f(xz[((size_t)(b * Lseq + ll)) * 768 + v * 384 + d]) * mconv_w[d * 3 + tt];
    }
  } else {    // reversed sequence: x[l]w2 + x[l+1]w1 + x[l+2]w0
    for (int u = 0; u < 3; u++) {
      int ll = l + u;
      if (ll < Lseq)
        acc += bf2f(xz[((size_t)(b * Lseq + ll)) * 768 + v * 384 + d]) * mconv_w[d * 3 + 2 - u];
    }
  }
  acc += mconv_b[d];
  xi_bf[i] = f2bf(siluf_(acc));
}

// ---------------------------------------------------------------------------
// dt projection (K=12, VALU) + softplus.
// ---------------------------------------------------------------------------
__global__ void dt_proj(const float* __restrict__ x_dbl,   // [4*M][48]
                        const float* __restrict__ dt_w,    // [192][12]
                        const float* __restrict__ dt_b,
                        float* __restrict__ dtv) {         // [4*M][192]
  size_t i = (size_t)blockIdx.x * blockDim.x + threadIdx.x;
  const size_t TOT = (size_t)4 * Mrow * Cch;
  if (i >= TOT) return;
  int d = (int)(i % Cch);
  size_t t = i / Cch;
  const float* xd = x_dbl + t * XDBL;
  float acc = dt_b[d];
  for (int r = 0; r < DTR; r++) acc = fmaf(xd[r], dt_w[d * DTR + r], acc);
  dtv[i] = softplusf_(acc);
}

// ---------------------------------------------------------------------------
// Selective scan. One (dir,b,d,n) state per lane; 16 lanes per d-row,
// 2 rows per wave32 (3072 rows -> 1536 waves). dA/dBx fused; y via shfl
// reduction; gate ( + D*xi ) * silu(z) stored bf16 for out_proj.
// ---------------------------------------------------------------------------
__global__ __launch_bounds__(256) void scan_k(const float* __restrict__ x_dbl,
                                              const float* __restrict__ dtv,
                                              const uint16_t* __restrict__ xi_bf,
                                              const uint16_t* __restrict__ xz,
                                              const float* __restrict__ A_log,
                                              const float* __restrict__ Dp,
                                              uint16_t* __restrict__ ygate) {
  int tid = threadIdx.x;
  int lane = tid & 31, wv = tid >> 5;
  int grp = lane >> 4, n = lane & 15;
  int row = (blockIdx.x * 8 + wv) * 2 + grp;     // 0..3071
  int d = row % Cch;
  int t = row / Cch;
  int b = t & 3, dir = t >> 2;
  int v = dir & 1, sf = dir >> 1;
  float And = -__expf(A_log[d * NST + n]);
  float Dd = Dp[d];
  float h = 0.f;
  const size_t mbase = (size_t)(dir * Mrow + b * Lseq);
  for (int step = 0; step < Lseq; step++) {
    int l = sf ? (Lseq - 1 - step) : step;
    size_t m = mbase + l;
    size_t mi = m * Cch + d;
    float dt_ = dtv[mi];
    float xi_ = bf2f(xi_bf[mi]);
    const float* xd = x_dbl + m * XDBL;
    float Bn = xd[DTR + n];
    float Cn = xd[DTR + NST + n];
    float dA = __expf(dt_ * And);
    float dBx = Bn * (dt_ * xi_);
    h = fmaf(dA, h, dBx);
    float y = h * Cn;
    y += __shfl_xor(y, 1, 32);
    y += __shfl_xor(y, 2, 32);
    y += __shfl_xor(y, 4, 32);
    y += __shfl_xor(y, 8, 32);
    if (n == 0) {
      float z = bf2f(xz[((size_t)(b * Lseq + l)) * 768 + v * 384 + Cch + d]);
      ygate[mi] = f2bf((y + Dd * xi_) * siluf_(z));
    }
  }
}

// ---------------------------------------------------------------------------
// Host launch
// ---------------------------------------------------------------------------
extern "C" void kernel_launch(void* const* d_in, const int* in_sizes, int n_in,
                              void* d_out, int out_size, void* d_ws, size_t ws_size,
                              hipStream_t stream) {
  const float* x       = (const float*)d_in[0];
  const float* conv_w  = (const float*)d_in[1];
  const float* conv_b  = (const float*)d_in[2];
  const float* eca_w   = (const float*)d_in[3];
  const float* in_w    = (const float*)d_in[4];
  const float* mconv_w = (const float*)d_in[5];
  const float* mconv_b = (const float*)d_in[6];
  const float* xproj_w = (const float*)d_in[7];
  const float* dt_w    = (const float*)d_in[8];
  const float* dt_b    = (const float*)d_in[9];
  const float* A_log   = (const float*)d_in[10];
  const float* Dp      = (const float*)d_in[11];
  const float* out_w   = (const float*)d_in[12];
  float* out = (float*)d_out;

  char* ws = (char*)d_ws;
  size_t off = 0;
  auto alloc = [&](size_t bytes) -> void* {
    void* p = ws + off;
    off = (off + bytes + 255) & ~(size_t)255;
    return p;
  };
  uint16_t* convW  = (uint16_t*)alloc((size_t)Cch * Cch * 2);
  uint16_t* inW    = (uint16_t*)alloc((size_t)768 * Cch * 2);
  uint16_t* xprojW = (uint16_t*)alloc((size_t)XDBL * Cch * 2);
  uint16_t* outW   = (uint16_t*)alloc((size_t)2 * Cch * Cch * 2);
  float*    pooled = (float*)alloc((size_t)Bsz * Cch * 4);
  float*    s_eca  = (float*)alloc((size_t)Bsz * Cch * 4);
  uint16_t* x_t    = (uint16_t*)alloc((size_t)Mrow * Cch * 2);
  uint16_t* x_flat = (uint16_t*)alloc((size_t)Mrow * Cch * 2);
  uint16_t* xz     = (uint16_t*)alloc((size_t)Mrow * 768 * 2);
  uint16_t* xi_bf  = (uint16_t*)alloc((size_t)4 * Mrow * Cch * 2);
  float*    x_dbl  = (float*)alloc((size_t)4 * Mrow * XDBL * 4);
  float*    dtv    = (float*)alloc((size_t)4 * Mrow * Cch * 4);
  uint16_t* ygate  = (uint16_t*)alloc((size_t)4 * Mrow * Cch * 2);

  prep_weights<<<64, 256, 0, stream>>>(conv_w, in_w, xproj_w, out_w, convW, inW, xprojW, outW);
  eca_pool<<<Bsz * Cch, 256, 0, stream>>>(x, pooled);
  eca_sig<<<3, 256, 0, stream>>>(pooled, eca_w, s_eca);
  transpose_bf16<<<dim3(Lseq / 32, Cch / 32, Bsz), dim3(32, 8), 0, stream>>>(x, x_t);

  {  // 1x1 conv: x_flat(B,L,C) = x_t @ convW^T + conv_b
    Segs4 s{};
    s.A[0] = x_t; s.B[0] = convW;
    EpiBiasBf e{conv_b, x_flat, Cch};
    gemm_bf16_k<4, 1, EpiBiasBf><<<dim3(Mrow / 64, Cch / 64), 128, 0, stream>>>(s, Cch, Cch, Cch, e);
  }
  {  // in_proj: both chan-flip variants as one N=768 GEMM
    Segs4 s{};
    s.A[0] = x_flat; s.B[0] = inW;
    EpiBf e{xz, 768};
    gemm_bf16_k<4, 1, EpiBf><<<dim3(Mrow / 64, 768 / 64), 128, 0, stream>>>(s, Cch, Cch, Cch, e);
  }
  {  // depthwise conv + silu -> xi (4 direction variants)
    size_t tot = (size_t)4 * Mrow * Cch;
    dwconv_silu<<<(unsigned)((tot + 255) / 256), 256, 0, stream>>>(xz, mconv_w, mconv_b, xi_bf);
  }
  {  // x_proj: all 4 dirs batched as M = 4*16384, N = 48
    Segs4 s{};
    s.A[0] = xi_bf; s.B[0] = xprojW;
    EpiF32 e{x_dbl, XDBL};
    gemm_bf16_k<3, 1, EpiF32><<<dim3(4 * Mrow / 64, 1), 128, 0, stream>>>(s, Cch, Cch, Cch, e);
  }
  {  // dt projection + softplus
    size_t tot = (size_t)4 * Mrow * Cch;
    dt_proj<<<(unsigned)((tot + 255) / 256), 256, 0, stream>>>(x_dbl, dt_w, dt_b, dtv);
  }
  // selective scan: 3072 rows, 2 per wave, 8 waves per block
  scan_k<<<192, 256, 0, stream>>>(x_dbl, dtv, xi_bf, xz, A_log, Dp, ygate);
  {  // out_proj: accumulate 4 directions; epilogue = /4 + ECA residual + shuffle
    Segs4 s{};
    for (int dir = 0; dir < 4; dir++) {
      s.A[dir] = ygate + (size_t)dir * Mrow * Cch;
      s.B[dir] = outW + (size_t)(dir & 1) * Cch * Cch;
    }
    EpiFinal e{x, s_eca, out};
    gemm_bf16_k<4, 4, EpiFinal><<<dim3(Mrow / 64, Cch / 64), 128, 0, stream>>>(s, Cch, Cch, Cch, e);
  }
  (void)in_sizes; (void)n_in; (void)out_size; (void)ws_size;
}
